// SelfAttention_57784490000998
// MI455X (gfx1250) — matrix operand order repaired and checked by reading, not compiled, not run on hardware
//
#include <hip/hip_runtime.h>
#include <hip/hip_bf16.h>
#include <stdint.h>

// ---------------------------------------------------------------------------
// Self-attention, B=2 S=4096 D=768 H=12 DV=64, fp32 I/O.
// All matmuls via v_wmma_f32_16x16x32_bf16 (bf16 A/B, f32 accum), wave32.
// x and all weight matrices are pre-converted to bf16 once, so every GEMM
// hot loop is pure b128 loads + WMMA (no per-tile f32->bf16 conversion).
// ---------------------------------------------------------------------------

constexpr int kB  = 2;
constexpr int kS  = 4096;
constexpr int kD  = 768;
constexpr int kH  = 12;
constexpr int kDV = 64;
constexpr int kTok = kB * kS;          // 8192 tokens

typedef __attribute__((ext_vector_type(16))) __bf16 bf16x16;
typedef __attribute__((ext_vector_type(8)))  __bf16 bf16x8;
typedef __attribute__((ext_vector_type(8)))  float  f32x8;
typedef __attribute__((ext_vector_type(4)))  float  f32x4;

static __device__ __forceinline__ f32x8 wmma_bf16(bf16x16 a, bf16x16 b, f32x8 c) {
  // 8 args: (neg_a, A, neg_b, B, c_mod, C, reuse_a, reuse_b)
  return __builtin_amdgcn_wmma_f32_16x16x32_bf16(false, a, false, b, (short)0, c,
                                                 false, false);
}

static __device__ __forceinline__ bf16x8 ldg8(const __bf16* p) {
  return *reinterpret_cast<const bf16x8*>(p);          // 16 B -> b128 load
}
static __device__ __forceinline__ f32x4 ldg4f(const float* p) {
  return *reinterpret_cast<const f32x4*>(p);           // 16 B -> b128 load
}
static __device__ __forceinline__ bf16x8 cvt8(f32x4 a, f32x4 b) {
  bf16x8 r;
  r[0] = (__bf16)a[0]; r[1] = (__bf16)a[1]; r[2] = (__bf16)a[2]; r[3] = (__bf16)a[3];
  r[4] = (__bf16)b[0]; r[5] = (__bf16)b[1]; r[6] = (__bf16)b[2]; r[7] = (__bf16)b[3];
  return r;
}
static __device__ __forceinline__ bf16x16 cat16(bf16x8 lo, bf16x8 hi) {
  return __builtin_shufflevector(lo, hi, 0, 1, 2, 3, 4, 5, 6, 7,
                                 8, 9, 10, 11, 12, 13, 14, 15);
}

// Layouts used here (wave32, same contiguous k order for A and B):
//  A 16x32 bf16 : lane lm = row M; elements = K run [16*hf, 16*hf+15]
//                 -> one contiguous 32 B chunk (two 16 B loads).
//  B 32x16 bf16 : lane lm = col N; elements = K run [16*hf, 16*hf+15]
//                 -> one contiguous 32 B chunk (two 16 B loads).
//  C 16x16 f32  : VGPR j holds row M = j + 8*hf, col N = lm.

// ---------------------------------------------------------------------------
// Kernel 0: one-pass f32 -> bf16 conversion (8 elements per thread).
// ---------------------------------------------------------------------------
__global__ __launch_bounds__(256)
void cvt_bf16_kernel(const float* __restrict__ src, uint16_t* __restrict__ dst,
                     int n8) {
  const int i = blockIdx.x * blockDim.x + threadIdx.x;
  if (i < n8) {
    const f32x4 a = ldg4f(src + (size_t)i * 8);
    const f32x4 b = ldg4f(src + (size_t)i * 8 + 4);
    *reinterpret_cast<bf16x8*>(reinterpret_cast<__bf16*>(dst) + (size_t)i * 8) =
        cvt8(a, b);
  }
}

// ---------------------------------------------------------------------------
// Kernel 1: QKV projection.  y = x @ W^T + b (all-bf16 fragments).
// Q,K stored bf16 [B,H,S,DV]; V stored TRANSPOSED bf16 [B,H,DV,S] so the
// attention P@V B-fragments are contiguous along the key dimension.
// grid = (512 token tiles, 12 64-wide N tiles == head, 3), block = 32.
// ---------------------------------------------------------------------------
__global__ __launch_bounds__(32)
void qkv_proj_kernel(const uint16_t* __restrict__ xraw,
                     const uint16_t* __restrict__ wqraw,
                     const uint16_t* __restrict__ wkraw,
                     const uint16_t* __restrict__ wvraw,
                     const float* __restrict__ bq, const float* __restrict__ bk,
                     const float* __restrict__ bv,
                     uint16_t* __restrict__ qraw, uint16_t* __restrict__ kraw,
                     uint16_t* __restrict__ vraw) {
  const int lane = threadIdx.x & 31;
  const int lm = lane & 15;
  const int hf = lane >> 4;
  const int rt = blockIdx.x;      // token tile (16 rows)
  const int nt = blockIdx.y;      // 64-wide feature tile == head index
  const int which = blockIdx.z;   // 0=Q 1=K 2=V

  const __bf16* X = reinterpret_cast<const __bf16*>(xraw);
  const __bf16* W = reinterpret_cast<const __bf16*>(
      (which == 0) ? wqraw : (which == 1) ? wkraw : wvraw);
  const float* bias = (which == 0) ? bq : (which == 1) ? bk : bv;
  __bf16* out = reinterpret_cast<__bf16*>((which == 0) ? qraw
                                          : (which == 1) ? kraw : vraw);
  const float scale = (which == 0) ? 0.125f : 1.0f;  // fold 1/sqrt(64) into Q

  const __bf16* xrow = X + (size_t)(rt * 16 + lm) * kD;
  const __bf16* wrow[4];
#pragma unroll
  for (int t = 0; t < 4; ++t)
    wrow[t] = W + (size_t)(nt * 64 + t * 16 + lm) * kD;

  f32x8 acc[4] = {};
  for (int k0 = 0; k0 < kD; k0 += 32) {
    const __bf16* ap = xrow + k0 + 16 * hf;
    const bf16x16 a = cat16(ldg8(ap), ldg8(ap + 8));
#pragma unroll
    for (int t = 0; t < 4; ++t) {
      const __bf16* wp = wrow[t] + k0 + 16 * hf;
      const bf16x16 b = cat16(ldg8(wp), ldg8(wp + 8));
      acc[t] = wmma_bf16(a, b, acc[t]);
    }
  }

  const int head = nt;
#pragma unroll
  for (int t = 0; t < 4; ++t) {
    const int dv = t * 16 + lm;
    const float bval = bias[nt * 64 + dv];
#pragma unroll
    for (int j = 0; j < 8; ++j) {          // C layout: row = j + 8*hf, col = lm
      const int token = rt * 16 + j + 8 * hf;
      const int bidx  = token >> 12;       // tiles never straddle the batch
      const int s     = token & 4095;
      const float val = (acc[t][j] + bval) * scale;
      const size_t base = ((size_t)bidx * kH + head) * ((size_t)kS * kDV);
      if (which == 2)
        out[base + (size_t)dv * kS + s] = (__bf16)val;   // V^T : [dv][key]
      else
        out[base + (size_t)s * kDV + dv] = (__bf16)val;  // Q,K : [tok][dv]
    }
  }
}

// ---------------------------------------------------------------------------
// Kernel 2: flash-style attention, one 16-query tile per wave.
// grid = (256, 12, 2), block = 32.  Online softmax over key tiles of 32.
// ---------------------------------------------------------------------------
__global__ __launch_bounds__(32)
void attn_kernel(const uint16_t* __restrict__ qraw,
                 const uint16_t* __restrict__ kraw,
                 const uint16_t* __restrict__ vraw,
                 uint16_t* __restrict__ oraw) {
  __shared__ alignas(16) __bf16 pbuf[16 * 32];  // P tile: C-layout -> A-layout

  const int lane = threadIdx.x & 31;
  const int lm = lane & 15;
  const int hf = lane >> 4;
  const int qt   = blockIdx.x;
  const int head = blockIdx.y;
  const int bidx = blockIdx.z;

  const size_t base = ((size_t)bidx * kH + head) * ((size_t)kS * kDV);
  const __bf16* Q  = reinterpret_cast<const __bf16*>(qraw) + base;
  const __bf16* K  = reinterpret_cast<const __bf16*>(kraw) + base;
  const __bf16* Vt = reinterpret_cast<const __bf16*>(vraw) + base; // [dv][key]
  __bf16* O = reinterpret_cast<__bf16*>(oraw);

  // Q A-fragments for DV=64 (two K=32 chunks), held in VGPRs for the loop.
  const __bf16* qrow = Q + (size_t)(qt * 16 + lm) * kDV;
  const bf16x16 aq0 = cat16(ldg8(qrow + 16 * hf), ldg8(qrow + 8 + 16 * hf));
  const bf16x16 aq1 = cat16(ldg8(qrow + 32 + 16 * hf), ldg8(qrow + 40 + 16 * hf));

  f32x8 o[4] = {};
  float m[8], l[8];
#pragma unroll
  for (int j = 0; j < 8; ++j) { m[j] = -3.0e38f; l[j] = 0.0f; }

  for (int kt = 0; kt < kS; kt += 32) {
    // prefetch next key/value tiles (global_prefetch_b8)
    if (kt + 32 < kS) {
      __builtin_prefetch(K + (size_t)(kt + 32 + lane) * kDV, 0, 3);
      __builtin_prefetch(Vt + (size_t)lane * kS + kt + 32, 0, 3);
      __builtin_prefetch(Vt + (size_t)(lane + 32) * kS + kt + 32, 0, 3);
    }

    // ---- scores: two 16-key subtiles, K-dim = DV = 64 = 2 x 32 ----
    f32x8 s0 = {}, s1 = {};
    const __bf16* kr0 = K + (size_t)(kt + lm) * kDV + 16 * hf;
    const __bf16* kr1 = K + (size_t)(kt + 16 + lm) * kDV + 16 * hf;
#pragma unroll
    for (int c = 0; c < 2; ++c) {
      const bf16x16 b0 = cat16(ldg8(kr0 + c * 32), ldg8(kr0 + c * 32 + 8));
      const bf16x16 b1 = cat16(ldg8(kr1 + c * 32), ldg8(kr1 + c * 32 + 8));
      s0 = wmma_bf16(c ? aq1 : aq0, b0, s0);
      s1 = wmma_bf16(c ? aq1 : aq0, b1, s1);
    }

    // ---- online softmax update (rows live inside 16-lane halves) ----
    float p0[8], p1[8];
#pragma unroll
    for (int j = 0; j < 8; ++j) {
      float t = fmaxf(s0[j], s1[j]);
      t = fmaxf(t, __shfl_xor(t, 1));
      t = fmaxf(t, __shfl_xor(t, 2));
      t = fmaxf(t, __shfl_xor(t, 4));
      t = fmaxf(t, __shfl_xor(t, 8));
      const float mn   = fmaxf(m[j], t);
      const float corr = __expf(m[j] - mn);
      p0[j] = __expf(s0[j] - mn);
      p1[j] = __expf(s1[j] - mn);
      float rs = p0[j] + p1[j];
      rs += __shfl_xor(rs, 1);
      rs += __shfl_xor(rs, 2);
      rs += __shfl_xor(rs, 4);
      rs += __shfl_xor(rs, 8);
      l[j] = l[j] * corr + rs;
      m[j] = mn;
#pragma unroll
      for (int f = 0; f < 4; ++f) o[f][j] = o[f][j] * corr;
    }

    // ---- transpose P (C layout) into A layout via LDS ----
#pragma unroll
    for (int j = 0; j < 8; ++j) {
      const int r = j + 8 * hf;
      pbuf[r * 32 + lm]      = (__bf16)p0[j];
      pbuf[r * 32 + 16 + lm] = (__bf16)p1[j];
    }
    // single wave per block: LDS ops from one wave are in-order
    const __bf16* pr = &pbuf[lm * 32 + 16 * hf];
    const bf16x16 pa = cat16(ldg8(pr), ldg8(pr + 8));  // 2x ds_load_b128

    // ---- O += P @ V   (V^T rows contiguous along key) ----
#pragma unroll
    for (int f = 0; f < 4; ++f) {
      const __bf16* vc = Vt + (size_t)(f * 16 + lm) * kS + kt + 16 * hf;
      const bf16x16 bv = cat16(ldg8(vc), ldg8(vc + 8));
      o[f] = wmma_bf16(pa, bv, o[f]);
    }
  }

  // ---- normalize and store attn output bf16 [tok][D], d = head*64 + dv ----
#pragma unroll
  for (int j = 0; j < 8; ++j) {
    const float inv = 1.0f / l[j];
    const int token = qt * 16 + j + 8 * hf;
    const size_t ro = ((size_t)bidx * kS + token) * kD + (size_t)head * kDV;
#pragma unroll
    for (int f = 0; f < 4; ++f)
      O[ro + f * 16 + lm] = (__bf16)(o[f][j] * inv);
  }
}

// ---------------------------------------------------------------------------
// Kernel 3: output projection.  y = attn @ Wo^T + bo  (f32 out).
// grid = (512, 12), block = 32; 16x64 output tile per wave, all-bf16 loads.
// ---------------------------------------------------------------------------
__global__ __launch_bounds__(32)
void out_proj_kernel(const uint16_t* __restrict__ araw,
                     const uint16_t* __restrict__ woraw,
                     const float* __restrict__ bo,
                     float* __restrict__ y) {
  const int lane = threadIdx.x & 31;
  const int lm = lane & 15;
  const int hf = lane >> 4;
  const int rt = blockIdx.x;
  const int nt = blockIdx.y;

  const __bf16* A  = reinterpret_cast<const __bf16*>(araw);
  const __bf16* Wo = reinterpret_cast<const __bf16*>(woraw);
  const __bf16* arow = A + (size_t)(rt * 16 + lm) * kD;
  const __bf16* wrow[4];
#pragma unroll
  for (int t = 0; t < 4; ++t)
    wrow[t] = Wo + (size_t)(nt * 64 + t * 16 + lm) * kD;

  f32x8 acc[4] = {};
  for (int k0 = 0; k0 < kD; k0 += 32) {
    const __bf16* ap = arow + k0 + 16 * hf;
    const bf16x16 a = cat16(ldg8(ap), ldg8(ap + 8));
#pragma unroll
    for (int t = 0; t < 4; ++t) {
      const __bf16* wp = wrow[t] + k0 + 16 * hf;
      const bf16x16 b = cat16(ldg8(wp), ldg8(wp + 8));
      acc[t] = wmma_bf16(a, b, acc[t]);
    }
  }

#pragma unroll
  for (int t = 0; t < 4; ++t) {
    const int ncol = nt * 64 + t * 16 + lm;
    const float bval = bo[ncol];
#pragma unroll
    for (int j = 0; j < 8; ++j) {
      const int token = rt * 16 + j + 8 * hf;
      y[(size_t)token * kD + ncol] = acc[t][j] + bval;
    }
  }
}

// ---------------------------------------------------------------------------
extern "C" void kernel_launch(void* const* d_in, const int* in_sizes, int n_in,
                              void* d_out, int out_size, void* d_ws,
                              size_t ws_size, hipStream_t stream) {
  const float* x  = (const float*)d_in[0];
  const float* Wq = (const float*)d_in[1];
  const float* bq = (const float*)d_in[2];
  const float* Wk = (const float*)d_in[3];
  const float* bk = (const float*)d_in[4];
  const float* Wv = (const float*)d_in[5];
  const float* bv = (const float*)d_in[6];
  const float* Wo = (const float*)d_in[7];
  const float* bo = (const float*)d_in[8];
  float* out = (float*)d_out;

  // workspace (bf16 elements):
  //   qb,kb : [B,H,S,DV]   vb : [B,H,DV,S]   ab : [tok,D]
  //   xb    : [tok,D]      wqb,wkb,wvb,wob : [D,D]
  const size_t tokD = (size_t)kTok * kD;   // 6,291,456
  const size_t DD   = (size_t)kD * kD;     //   589,824
  uint16_t* qb  = (uint16_t*)d_ws;
  uint16_t* kb  = qb + tokD;
  uint16_t* vb  = kb + tokD;
  uint16_t* ab  = vb + tokD;
  uint16_t* xb  = ab + tokD;
  uint16_t* wqb = xb + tokD;
  uint16_t* wkb = wqb + DD;
  uint16_t* wvb = wkb + DD;
  uint16_t* wob = wvb + DD;

  const dim3 blk(32);
  // one-pass f32 -> bf16 materialization (amortized over 512 tile reuses)
  cvt_bf16_kernel<<<dim3((int)(tokD / 8 + 255) / 256), 256, 0, stream>>>(
      x, xb, (int)(tokD / 8));
  cvt_bf16_kernel<<<dim3((int)(DD / 8 + 255) / 256), 256, 0, stream>>>(
      Wq, wqb, (int)(DD / 8));
  cvt_bf16_kernel<<<dim3((int)(DD / 8 + 255) / 256), 256, 0, stream>>>(
      Wk, wkb, (int)(DD / 8));
  cvt_bf16_kernel<<<dim3((int)(DD / 8 + 255) / 256), 256, 0, stream>>>(
      Wv, wvb, (int)(DD / 8));
  cvt_bf16_kernel<<<dim3((int)(DD / 8 + 255) / 256), 256, 0, stream>>>(
      Wo, wob, (int)(DD / 8));

  qkv_proj_kernel<<<dim3(kTok / 16, kH, 3), blk, 0, stream>>>(
      xb, wqb, wkb, wvb, bq, bk, bv, qb, kb, vb);
  attn_kernel<<<dim3(kS / 16, kH, kB), blk, 0, stream>>>(qb, kb, vb, ab);
  out_proj_kernel<<<dim3(kTok / 16, kD / 64), blk, 0, stream>>>(ab, wob, bo, out);
}
